// LlamaMLP_13683765805223
// MI455X (gfx1250) — 8-bit matrix operand order repaired and checked by reading, compiled, not run on hardware
//
#include <hip/hip_runtime.h>
#include <hip/hip_bf16.h>
#include <math.h>

#define T_DIM 4096
#define H_DIM 4096
#define I_DIM 11008

#define PF_DIST_S "1024"

typedef __attribute__((ext_vector_type(8))) int v8i;

__device__ __forceinline__ v8i wmma_iu8(v8i a, v8i b, v8i c) {
  return __builtin_amdgcn_wmma_i32_16x16x64_iu8(true, a, true, b, c, false, false);
}

__device__ __forceinline__ void prefetch_wgp(const char* p) {
  asm volatile("global_prefetch_b8 %0, off offset:" PF_DIST_S ::"v"(p));
}

__device__ __forceinline__ int pack4(int4 v) {
  return (v.x & 0xFF) | ((v.y & 0xFF) << 8) | ((v.z & 0xFF) << 16) | ((v.w & 0xFF) << 24);
}

__global__ void pack_i32_to_i8(const int4* __restrict__ src,
                               int4* __restrict__ dst, int n16) {
  int idx = blockIdx.x * blockDim.x + threadIdx.x;
  if (idx >= n16) return;
  int4 o;
  o.x = pack4(src[idx * 4 + 0]);
  o.y = pack4(src[idx * 4 + 1]);
  o.z = pack4(src[idx * 4 + 2]);
  o.w = pack4(src[idx * 4 + 3]);
  dst[idx] = o;
}

__device__ __forceinline__ v8i load_a_frag(const char* p) {
  const int4 t0 = *(const int4*)(p + 0);
  const int4 t1 = *(const int4*)(p + 32);
  v8i a;
  a[0] = t0.x; a[1] = t0.y; a[2] = t0.z; a[3] = t0.w;
  a[4] = t1.x; a[5] = t1.y; a[6] = t1.z; a[7] = t1.w;
  return a;
}

__device__ __forceinline__ v8i load_b_frag(const char* p) {
  const int4 u0 = *(const int4*)(p + 0);
  const int4 u1 = *(const int4*)(p + 32);
  v8i b;
  b[0] = u0.x; b[1] = u0.y; b[2] = u0.z; b[3] = u0.w;
  b[4] = u1.x; b[5] = u1.y; b[6] = u1.z; b[7] = u1.w;
  return b;
}

__device__ __forceinline__ float fast_silu_mul(float g, float u) {
  const float sig = __builtin_amdgcn_rcpf(1.0f + __expf(-g));
  return g * sig * u;
}

__global__ __launch_bounds__(256)
void gemm_gu_silu(const char* __restrict__ x8, const char* __restrict__ wgu8,
                  const float* __restrict__ x_scale, const float* __restrict__ s_gu,
                  float* __restrict__ y, unsigned int* __restrict__ rowmax_bits) {
  const int lane = threadIdx.x & 31;
  const int wave = threadIdx.x >> 5;
  const int ksel = lane >> 4;
  const int r    = lane & 15;

  const int nGroup = blockIdx.x;
  const int mGroup = blockIdx.y * 8 + wave;
  const int m0  = mGroup * 32;
  const int nG0 = nGroup * 32;
  const int nU0 = I_DIM + nG0;

  const char* a0p  = x8   + (size_t)(m0 + r)  * H_DIM + ksel * 16;
  const char* a1p  = a0p  + (size_t)16 * H_DIM;
  const char* bg0p = wgu8 + (size_t)(nG0 + r) * H_DIM + ksel * 16;
  const char* bg1p = bg0p + (size_t)16 * H_DIM;
  const char* bu0p = wgu8 + (size_t)(nU0 + r) * H_DIM + ksel * 16;
  const char* bu1p = bu0p + (size_t)16 * H_DIM;

  v8i cg[2][2] = {};
  v8i cu[2][2] = {};

  auto kstep = [&](int kk) {
    const v8i a0  = load_a_frag(a0p + kk);
    const v8i a1  = load_a_frag(a1p + kk);
    const v8i bg0 = load_b_frag(bg0p + kk);
    const v8i bg1 = load_b_frag(bg1p + kk);
    const v8i bu0 = load_b_frag(bu0p + kk);
    const v8i bu1 = load_b_frag(bu1p + kk);
    cg[0][0] = wmma_iu8(a0, bg0, cg[0][0]);
    cg[0][1] = wmma_iu8(a0, bg1, cg[0][1]);
    cg[1][0] = wmma_iu8(a1, bg0, cg[1][0]);
    cg[1][1] = wmma_iu8(a1, bg1, cg[1][1]);
    cu[0][0] = wmma_iu8(a0, bu0, cu[0][0]);
    cu[0][1] = wmma_iu8(a0, bu1, cu[0][1]);
    cu[1][0] = wmma_iu8(a1, bu0, cu[1][0]);
    cu[1][1] = wmma_iu8(a1, bu1, cu[1][1]);
  };

  for (int k0 = 0; k0 < H_DIM; k0 += 128) {
    prefetch_wgp(a0p + k0);
    prefetch_wgp(bg0p + k0);
    prefetch_wgp(bu0p + k0);
    kstep(k0);
    kstep(k0 + 64);
  }

  const float sg0 = s_gu[nG0 + r];
  const float sg1 = s_gu[nG0 + 16 + r];
  const float su0 = s_gu[nU0 + r];
  const float su1 = s_gu[nU0 + 16 + r];

#pragma unroll
  for (int ms = 0; ms < 2; ++ms) {
#pragma unroll
    for (int j = 0; j < 8; ++j) {
      const int m = m0 + ms * 16 + j + ksel * 8;
      const float xs = x_scale[m];
      float av = 0.0f;
#pragma unroll
      for (int ns = 0; ns < 2; ++ns) {
        const float g = (float)cg[ms][ns][j] * xs * (ns ? sg1 : sg0);
        const float u = (float)cu[ms][ns][j] * xs * (ns ? su1 : su0);
        const float yv = fast_silu_mul(g, u);
        y[(size_t)m * I_DIM + nG0 + ns * 16 + r] = yv;
        av = fmaxf(av, fabsf(yv));
      }
      av = fmaxf(av, __shfl_xor(av, 1));
      av = fmaxf(av, __shfl_xor(av, 2));
      av = fmaxf(av, __shfl_xor(av, 4));
      av = fmaxf(av, __shfl_xor(av, 8));
      if (r == 0)
        atomicMax(&rowmax_bits[m], __float_as_uint(av));
    }
  }
}

__global__ void quantize_rows(const float* __restrict__ y,
                              const unsigned int* __restrict__ rowmax_bits,
                              float* __restrict__ s2_out,
                              unsigned int* __restrict__ yq8, int total4) {
  int idx = blockIdx.x * blockDim.x + threadIdx.x;
  if (idx >= total4) return;
  const int cols4 = I_DIM / 4;
  const int row = idx / cols4;
  const float mx = fmaxf(__uint_as_float(rowmax_bits[row]), 1e-8f);
  const float s2 = mx * (1.0f / 127.0f);
  const float inv = 127.0f * __builtin_amdgcn_rcpf(mx);
  if (idx - row * cols4 == 0) s2_out[row] = s2;

  const float4 v = ((const float4*)y)[idx];
  int q0 = min(127, max(-128, (int)rintf(v.x * inv)));
  int q1 = min(127, max(-128, (int)rintf(v.y * inv)));
  int q2 = min(127, max(-128, (int)rintf(v.z * inv)));
  int q3 = min(127, max(-128, (int)rintf(v.w * inv)));
  yq8[idx] = (unsigned int)(q0 & 0xFF) |
             ((unsigned int)(q1 & 0xFF) << 8) |
             ((unsigned int)(q2 & 0xFF) << 16) |
             ((unsigned int)(q3 & 0xFF) << 24);
}

__global__ __launch_bounds__(256)
void gemm_down(const char* __restrict__ yq8, const char* __restrict__ wd8,
               const float* __restrict__ s2, const float* __restrict__ s_wd,
               float* __restrict__ out) {
  const int lane = threadIdx.x & 31;
  const int wave = threadIdx.x >> 5;
  const int ksel = lane >> 4;
  const int r    = lane & 15;

  const int nGroup = blockIdx.x;
  const int mGroup = blockIdx.y * 8 + wave;
  const int m0 = mGroup * 32;
  const int n0 = nGroup * 32;

  const char* a0p = yq8 + (size_t)(m0 + r) * I_DIM + ksel * 16;
  const char* a1p = a0p + (size_t)16 * I_DIM;
  const char* b0p = wd8 + (size_t)(n0 + r) * I_DIM + ksel * 16;
  const char* b1p = b0p + (size_t)16 * I_DIM;

  v8i c[2][2] = {};

  auto kstep = [&](int kk) {
    const v8i a0 = load_a_frag(a0p + kk);
    const v8i a1 = load_a_frag(a1p + kk);
    const v8i b0 = load_b_frag(b0p + kk);
    const v8i b1 = load_b_frag(b1p + kk);
    c[0][0] = wmma_iu8(a0, b0, c[0][0]);
    c[0][1] = wmma_iu8(a0, b1, c[0][1]);
    c[1][0] = wmma_iu8(a1, b0, c[1][0]);
    c[1][1] = wmma_iu8(a1, b1, c[1][1]);
  };

  for (int k0 = 0; k0 < I_DIM; k0 += 128) {
    prefetch_wgp(a0p + k0);
    prefetch_wgp(b0p + k0);
    kstep(k0);
    kstep(k0 + 64);
  }

  const float sw0 = s_wd[n0 + r];
  const float sw1 = s_wd[n0 + 16 + r];

#pragma unroll
  for (int ms = 0; ms < 2; ++ms) {
#pragma unroll
    for (int j = 0; j < 8; ++j) {
      const int m = m0 + ms * 16 + j + ksel * 8;
      const float sm = s2[m];
      out[(size_t)m * H_DIM + n0 + r]      = (float)c[ms][0][j] * sm * sw0;
      out[(size_t)m * H_DIM + n0 + 16 + r] = (float)c[ms][1][j] * sm * sw1;
    }
  }
}

extern "C" void kernel_launch(void* const* d_in, const int* in_sizes, int n_in,
                              void* d_out, int out_size, void* d_ws, size_t ws_size,
                              hipStream_t stream) {
  const int*   x_q     = (const int*)d_in[0];
  const float* x_scale = (const float*)d_in[1];
  const int*   w_gu    = (const int*)d_in[2];
  const float* s_gu    = (const float*)d_in[3];
  const int*   w_d     = (const int*)d_in[4];
  const float* s_wd    = (const float*)d_in[5];
  float* out = (float*)d_out;

  char* ws   = (char*)d_ws;
  char* x8   = ws;
  char* wgu8 = x8   + (size_t)T_DIM * H_DIM;
  char* wd8  = wgu8 + (size_t)2 * I_DIM * H_DIM;
  float* y   = (float*)(wd8 + (size_t)H_DIM * I_DIM);
  char* yq8  = (char*)(y + (size_t)T_DIM * I_DIM);
  unsigned int* rowmax = (unsigned int*)(yq8 + (size_t)T_DIM * I_DIM);
  float* s2  = (float*)(rowmax + T_DIM);

  hipMemsetAsync(rowmax, 0, T_DIM * sizeof(unsigned int), stream);

  int n16;
  n16 = T_DIM * H_DIM / 16;
  pack_i32_to_i8<<<(n16 + 255) / 256, 256, 0, stream>>>((const int4*)x_q, (int4*)x8, n16);
  n16 = 2 * I_DIM * H_DIM / 16;
  pack_i32_to_i8<<<(n16 + 255) / 256, 256, 0, stream>>>((const int4*)w_gu, (int4*)wgu8, n16);
  n16 = H_DIM * I_DIM / 16;
  pack_i32_to_i8<<<(n16 + 255) / 256, 256, 0, stream>>>((const int4*)w_d, (int4*)wd8, n16);

  gemm_gu_silu<<<dim3(I_DIM / 32, (T_DIM / 32) / 8), 256, 0, stream>>>(
      x8, wgu8, x_scale, s_gu, y, rowmax);

  int t4 = T_DIM * (I_DIM / 4);
  quantize_rows<<<(t4 + 255) / 256, 256, 0, stream>>>(
      y, rowmax, s2, (unsigned int*)yq8, t4);

  gemm_down<<<dim3(H_DIM / 32, (T_DIM / 32) / 8), 256, 0, stream>>>(
      yq8, wd8, s2, s_wd, out);
}
